// HybridCNNTransformerDA_18245021073949
// MI455X (gfx1250) — matrix operand order repaired and checked by reading, not compiled, not run on hardware
//
#include <hip/hip_runtime.h>
#include <hip/hip_bf16.h>
#include <math.h>

// ---------------------------------------------------------------------------
// Types for CDNA5 WMMA (wave32, 16x16x32 bf16 -> f32)
// ---------------------------------------------------------------------------
typedef __bf16 bf16_t;
typedef __attribute__((ext_vector_type(16))) __bf16 v16bf;
typedef __attribute__((ext_vector_type(8)))  float  v8f;

#define DEV __device__ __forceinline__

DEV float bf2f(bf16_t b) {
    unsigned short h = __builtin_bit_cast(unsigned short, b);
    unsigned u = ((unsigned)h) << 16;
    return __builtin_bit_cast(float, u);
}
DEV bf16_t f2bf(float f) {
    unsigned u = __builtin_bit_cast(unsigned, f);
    unsigned r = u + 0x7FFFu + ((u >> 16) & 1u);   // round-to-nearest-even
    unsigned short h = (unsigned short)(r >> 16);
    return __builtin_bit_cast(bf16_t, h);
}

union Frag16 { v16bf v; uint4 q[2]; };

// A-matrix fragment (16xK tile, row-major src, stride in elements, mult of 32)
// lane L in [0,15]: row M=L, 16 contiguous k's [kk*32+half*16, +16)
DEV v16bf load_frag_A(const bf16_t* base, int stride, int m_base, int kk, int lane) {
    int half = lane >> 4;
    int m = m_base + (lane & 15);
    const bf16_t* p = base + (size_t)m * stride + kk * 32 + half * 16;
    Frag16 f;
    f.q[0] = *(const uint4*)p;
    f.q[1] = *(const uint4*)(p + 8);
    return f.v;
}

// B-matrix fragment: B[k][n] = W[n][k] (W row-major NxK). lane holds column n,
// 16 contiguous k's starting at kk*32 + half*16 -> one 32B load (2x b128).
DEV v16bf load_frag_B(const bf16_t* W, int stride, int n, int kk, int lane) {
    int half = lane >> 4;
    const bf16_t* p = W + (size_t)n * stride + kk * 32 + half * 16;
    Frag16 f;
    f.q[0] = *(const uint4*)p;
    f.q[1] = *(const uint4*)(p + 8);
    return f.v;
}

DEV v8f wmma_bf16(v16bf a, v16bf b, v8f c) {
    return __builtin_amdgcn_wmma_f32_16x16x32_bf16(false, a, false, b, (short)0, c,
                                                   false, false);
}

DEV float eluf(float x)  { return x > 0.f ? x : expm1f(x); }
DEV float geluf(float x) { return 0.5f * x * (1.f + erff(x * 0.70710678118654752f)); }

// ---------------------------------------------------------------------------
// Problem constants & workspace layout
// ---------------------------------------------------------------------------
// B=512 C=129 T=900 D=128 H=8 NL=4
constexpr size_t F_H1  = 0;                            // 512*32*900 f32 (later h3)
constexpr size_t F_H2  = F_H1 + (size_t)512*32*900;    // 512*64*225 f32 (later h4)
constexpr size_t F_TA  = F_H2 + (size_t)512*64*225;    // 512*28*128 f32
constexpr size_t F_TB  = F_TA + (size_t)512*28*128;    // 512*28*128 f32
constexpr size_t F_ERP = F_TB + (size_t)512*28*128;    // 512*13 (padded 16)
constexpr size_t F_END = F_ERP + (size_t)512*16;

// bf16 region (element offsets from bf16 base)
constexpr size_t B_FEAT = 0;                            // 512*3616
constexpr size_t B_Y1   = B_FEAT + (size_t)512*3616;    // 512*512
constexpr size_t B_Y2   = B_Y1  + (size_t)512*512;      // 512*256
constexpr size_t B_Y3   = B_Y2  + (size_t)512*256;      // 512*128
constexpr size_t B_TMW  = B_Y3  + (size_t)512*128;      // 64*800
constexpr size_t B_LW   = B_TMW + (size_t)64*800;       // 4 layers * 196608
constexpr size_t LW_STRIDE = 49152 + 16384 + 65536 + 65536; // in,out,ff1,ff2
constexpr size_t B_W1   = B_LW  + 4*LW_STRIDE;          // 512*3616
constexpr size_t B_W2   = B_W1  + (size_t)512*3616;     // 256*512
constexpr size_t B_W3   = B_W2  + (size_t)256*512;      // 128*256

// ---------------------------------------------------------------------------
// Weight conversion
// ---------------------------------------------------------------------------
__global__ void k_cvt(const float* s, bf16_t* d, int n) {
    for (int i = blockIdx.x * 256 + threadIdx.x; i < n; i += gridDim.x * 256)
        d[i] = f2bf(s[i]);
}
__global__ void k_cvt_pad(const float* s, bf16_t* d, int rows, int K, int Kp) {
    int n = rows * Kp;
    for (int i = blockIdx.x * 256 + threadIdx.x; i < n; i += gridDim.x * 256) {
        int r = i / Kp, k = i - r * Kp;
        d[i] = (k < K) ? f2bf(s[(size_t)r * K + k]) : f2bf(0.f);
    }
}

// ---------------------------------------------------------------------------
// Stage 1: h1[b,o,t] = elu(bn1(sum_c x[b,c,t]*sp_w[o,c]))   (32x129 proj)
// ---------------------------------------------------------------------------
__global__ void __launch_bounds__(256) k_spatial(const float* __restrict__ x,
        const float* __restrict__ sw, const float* g, const float* bb,
        const float* m, const float* v, float* __restrict__ h1) {
    __shared__ float xt[129][65];
    int b = blockIdx.y, t0 = blockIdx.x * 64;
    for (int i = threadIdx.x; i < 129 * 64; i += 256) {
        int c = i >> 6, tl = i & 63, t = t0 + tl;
        xt[c][tl] = (t < 900) ? x[((size_t)b * 129 + c) * 900 + t] : 0.f;
    }
    __syncthreads();
    int o = threadIdx.x >> 3;
    int tb = threadIdx.x & 7;
    float acc[8];
    for (int j = 0; j < 8; j++) acc[j] = 0.f;
    for (int c = 0; c < 129; c++) {
        float w = sw[o * 129 + c];
        for (int j = 0; j < 8; j++) acc[j] += w * xt[c][tb * 8 + j];
    }
    float sc = g[o] * rsqrtf(v[o] + 1e-5f);
    float bi = bb[o] - m[o] * sc;
    for (int j = 0; j < 8; j++) {
        int t = t0 + tb * 8 + j;
        if (t < 900) h1[((size_t)b * 32 + o) * 900 + t] = eluf(acc[j] * sc + bi);
    }
}

// ---------------------------------------------------------------------------
// Stage 2: temporal conv 32->64 k=25 pad=12 as WMMA implicit GEMM (K=800),
// then bn2+elu+avgpool4 -> h2 (512,64,225)
// ---------------------------------------------------------------------------
__global__ void __launch_bounds__(256) k_temporal(const float* __restrict__ h1,
        const bf16_t* __restrict__ wbf, const float* g, const float* bb,
        const float* m, const float* v, float* __restrict__ h2) {
    __shared__ __align__(16) bf16_t hb[32][64];   // input tile, t0-12..t0+43
    __shared__ float pool[64][33];
    int b = blockIdx.y, t0 = blockIdx.x * 32;
    for (int i = threadIdx.x; i < 32 * 56; i += 256) {
        int c = i / 56, tl = i - c * 56;
        int t = t0 - 12 + tl;
        float val = (t >= 0 && t < 900) ? h1[((size_t)b * 32 + c) * 900 + t] : 0.f;
        hb[c][tl] = f2bf(val);
    }
    __syncthreads();
    int wave = threadIdx.x >> 5, lane = threadIdx.x & 31;
    int half = lane >> 4, ln = lane & 15;
    int mt = wave & 3, nt = wave >> 2;  // 4 M-tiles (64 ch) x 2 N-tiles (32 t)
    v8f acc = {};
    for (int kk = 0; kk < 25; kk++) {
        v16bf a = load_frag_A(wbf, 800, mt * 16, kk, lane);
        v16bf bv;
        int kbase = kk * 32 + half * 16;
        for (int i = 0; i < 16; i++) {
            int k = kbase + i;
            int c = k / 25, kt = k - c * 25;
            bv[i] = hb[c][nt * 16 + ln + kt];
        }
        acc = wmma_bf16(a, bv, acc);
    }
    for (int r = 0; r < 8; r++) {
        int o = mt * 16 + half * 8 + r;
        float sc = g[o] * rsqrtf(v[o] + 1e-5f);
        float bi = bb[o] - m[o] * sc;
        pool[o][nt * 16 + ln] = eluf(acc[r] * sc + bi);
    }
    __syncthreads();
    for (int i = threadIdx.x; i < 64 * 8; i += 256) {
        int o = i >> 3, j = i & 7;
        int p = blockIdx.x * 8 + j;
        if (p < 225) {
            float s4 = 0.25f * (pool[o][j*4] + pool[o][j*4+1] + pool[o][j*4+2] + pool[o][j*4+3]);
            h2[((size_t)b * 64 + o) * 225 + p] = s4;
        }
    }
}

// ---------------------------------------------------------------------------
// Separable conv block 1: (512,64,225) -> dw5 -> pw 64->128 -> bn elu -> pool2
// ---------------------------------------------------------------------------
__global__ void __launch_bounds__(256) k_sc1(const float* __restrict__ h2,
        const float* dw, const float* pw, const float* g, const float* bb,
        const float* m, const float* v, float* __restrict__ h3) {
    __shared__ float in[64][40];
    __shared__ float dwo[64][33];
    int b = blockIdx.y, t0 = blockIdx.x * 32;
    for (int i = threadIdx.x; i < 64 * 36; i += 256) {
        int c = i / 36, tl = i - c * 36;
        int t = t0 - 2 + tl;
        in[c][tl] = (t >= 0 && t < 225) ? h2[((size_t)b * 64 + c) * 225 + t] : 0.f;
    }
    __syncthreads();
    for (int i = threadIdx.x; i < 64 * 32; i += 256) {
        int c = i >> 5, tl = i & 31;
        float s = 0.f;
        for (int k = 0; k < 5; k++) s += in[c][tl + k] * dw[c * 5 + k];
        dwo[c][tl] = s;
    }
    __syncthreads();
    int o = threadIdx.x & 127, hseg = threadIdx.x >> 7;
    float sc = g[o] * rsqrtf(v[o] + 1e-5f);
    float bi = bb[o] - m[o] * sc;
    for (int j = 0; j < 8; j++) {
        int p = hseg * 8 + j;
        float s0 = 0.f, s1 = 0.f;
        for (int c = 0; c < 64; c++) {
            float w = pw[o * 64 + c];
            s0 += w * dwo[c][2 * p];
            s1 += w * dwo[c][2 * p + 1];
        }
        s0 = eluf(s0 * sc + bi);
        s1 = eluf(s1 * sc + bi);
        h3[((size_t)b * 128 + o) * 112 + blockIdx.x * 16 + p] = 0.5f * (s0 + s1);
    }
}

// ---------------------------------------------------------------------------
// Separable conv block 2: (512,128,112) -> dw5 -> pw 128->128 -> bn elu -> pool2
// ---------------------------------------------------------------------------
__global__ void __launch_bounds__(256) k_sc2(const float* __restrict__ h3,
        const float* dw, const float* pw, const float* g, const float* bb,
        const float* m, const float* v, float* __restrict__ h4) {
    __shared__ float in[128][21];
    __shared__ float dwo[128][17];
    int b = blockIdx.y, t0 = blockIdx.x * 16;
    for (int i = threadIdx.x; i < 128 * 20; i += 256) {
        int c = i / 20, tl = i - c * 20;
        int t = t0 - 2 + tl;
        in[c][tl] = (t >= 0 && t < 112) ? h3[((size_t)b * 128 + c) * 112 + t] : 0.f;
    }
    __syncthreads();
    for (int i = threadIdx.x; i < 128 * 16; i += 256) {
        int c = i >> 4, tl = i & 15;
        float s = 0.f;
        for (int k = 0; k < 5; k++) s += in[c][tl + k] * dw[c * 5 + k];
        dwo[c][tl] = s;
    }
    __syncthreads();
    int o = threadIdx.x & 127, hseg = threadIdx.x >> 7;
    float sc = g[o] * rsqrtf(v[o] + 1e-5f);
    float bi = bb[o] - m[o] * sc;
    for (int j = 0; j < 4; j++) {
        int p = hseg * 4 + j;
        float s0 = 0.f, s1 = 0.f;
        for (int c = 0; c < 128; c++) {
            float w = pw[o * 128 + c];
            s0 += w * dwo[c][2 * p];
            s1 += w * dwo[c][2 * p + 1];
        }
        s0 = eluf(s0 * sc + bi);
        s1 = eluf(s1 * sc + bi);
        h4[((size_t)b * 128 + o) * 56 + blockIdx.x * 8 + p] = 0.5f * (s0 + s1);
    }
}

// ---------------------------------------------------------------------------
// pool2(56->28) + transpose + positional encoding -> tA (512,28,128)
// ---------------------------------------------------------------------------
__global__ void k_pack_t(const float* __restrict__ h4, float* __restrict__ tA) {
    int b = blockIdx.x;
    for (int i = threadIdx.x; i < 28 * 128; i += 256) {
        int s = i >> 7, d = i & 127;
        float val = 0.5f * (h4[((size_t)b * 128 + d) * 56 + 2 * s] +
                            h4[((size_t)b * 128 + d) * 56 + 2 * s + 1]);
        int ip = d >> 1;
        float div = expf((float)(2 * ip) * (-9.2103403719761836f / 128.f));
        float pe = (d & 1) ? cosf((float)s * div) : sinf((float)s * div);
        tA[((size_t)b * 28 + s) * 128 + d] = val + pe;
    }
}

// ---------------------------------------------------------------------------
// Transformer attention block: tout = LN1(tin + proj(attn(QKV(tin))))
// Per-block batch element. WMMA for QKV (32x384,K=128) and out-proj (32x128).
// ---------------------------------------------------------------------------
__global__ void __launch_bounds__(256) k_attn(const float* __restrict__ tin,
        const bf16_t* __restrict__ wqkv, const float* bqkv,
        const bf16_t* __restrict__ wout, const float* bout,
        const float* lng, const float* lnb, float* __restrict__ tout) {
    __shared__ __align__(16) bf16_t abf[32][128];
    __shared__ __align__(16) bf16_t qkv[32][384];
    __shared__ __align__(16) bf16_t obf[32][128];
    __shared__ float proj[32][128];
    int b = blockIdx.x;
    const float* tb = tin + (size_t)b * 28 * 128;
    for (int i = threadIdx.x; i < 32 * 128; i += 256) {
        int r = i >> 7, c = i & 127;
        abf[r][c] = f2bf(r < 28 ? tb[r * 128 + c] : 0.f);
    }
    __syncthreads();
    int wave = threadIdx.x >> 5, lane = threadIdx.x & 31;
    int half = lane >> 4, ln = lane & 15;
    // QKV: 2 M-tiles x 24 N-tiles, K = 128 (4 steps)
    for (int tile = wave; tile < 48; tile += 8) {
        int mt = tile / 24, nt = tile - mt * 24;
        v8f acc = {};
        for (int kk = 0; kk < 4; kk++) {
            v16bf a  = load_frag_A(&abf[0][0], 128, mt * 16, kk, lane);
            v16bf bv = load_frag_B(wqkv, 128, nt * 16 + ln, kk, lane);
            acc = wmma_bf16(a, bv, acc);
        }
        int n = nt * 16 + ln;
        float bias = bqkv[n];
        for (int r = 0; r < 8; r++)
            qkv[mt * 16 + half * 8 + r][n] = f2bf(acc[r] + bias);
    }
    __syncthreads();
    // attention: wave == head (Dh=16), lane == query row
    {
        int h = wave, r = lane;
        if (r < 28) {
            float scor[28];
            float mx = -1e30f;
            for (int kc = 0; kc < 28; kc++) {
                float s = 0.f;
                for (int d = 0; d < 16; d++)
                    s += bf2f(qkv[r][h * 16 + d]) * bf2f(qkv[kc][128 + h * 16 + d]);
                s *= 0.25f;          // 1/sqrt(16)
                scor[kc] = s;
                if (s > mx) mx = s;
            }
            float den = 0.f;
            for (int kc = 0; kc < 28; kc++) { scor[kc] = expf(scor[kc] - mx); den += scor[kc]; }
            float inv = 1.f / den;
            for (int d = 0; d < 16; d++) {
                float o = 0.f;
                for (int kc = 0; kc < 28; kc++)
                    o += scor[kc] * bf2f(qkv[kc][256 + h * 16 + d]);
                obf[r][h * 16 + d] = f2bf(o * inv);
            }
        } else {
            for (int d = 0; d < 16; d++) obf[r][h * 16 + d] = f2bf(0.f);
        }
    }
    __syncthreads();
    // out projection: 2 x 8 tiles, K = 128
    for (int tile = wave; tile < 16; tile += 8) {
        int mt = tile >> 3, nt = tile & 7;
        v8f acc = {};
        for (int kk = 0; kk < 4; kk++) {
            v16bf a  = load_frag_A(&obf[0][0], 128, mt * 16, kk, lane);
            v16bf bv = load_frag_B(wout, 128, nt * 16 + ln, kk, lane);
            acc = wmma_bf16(a, bv, acc);
        }
        int n = nt * 16 + ln;
        float bias = bout[n];
        for (int r = 0; r < 8; r++) proj[mt * 16 + half * 8 + r][n] = acc[r] + bias;
    }
    __syncthreads();
    // residual + LayerNorm (wave-per-row, shfl reduce over 32 lanes)
    for (int r = wave; r < 28; r += 8) {
        float vals[4];
        float s = 0.f;
        for (int j = 0; j < 4; j++) {
            int c = lane * 4 + j;
            vals[j] = tb[r * 128 + c] + proj[r][c];
            s += vals[j];
        }
        for (int off = 16; off; off >>= 1) s += __shfl_xor(s, off, 32);
        float mean = s * (1.f / 128.f);
        float vs = 0.f;
        for (int j = 0; j < 4; j++) { float d = vals[j] - mean; vs += d * d; }
        for (int off = 16; off; off >>= 1) vs += __shfl_xor(vs, off, 32);
        float inv = rsqrtf(vs * (1.f / 128.f) + 1e-5f);
        float* outr = tout + (size_t)b * 28 * 128 + r * 128;
        for (int j = 0; j < 4; j++) {
            int c = lane * 4 + j;
            outr[c] = (vals[j] - mean) * inv * lng[c] + lnb[c];
        }
    }
}

// ---------------------------------------------------------------------------
// Transformer FF block: tout = LN2(tin + W2(gelu(W1 tin + b1)) + b2)
// ---------------------------------------------------------------------------
__global__ void __launch_bounds__(256) k_ffn(const float* __restrict__ tin,
        const bf16_t* __restrict__ w1, const float* b1,
        const bf16_t* __restrict__ w2, const float* b2,
        const float* lng, const float* lnb, float* __restrict__ tout) {
    __shared__ __align__(16) bf16_t abf[32][128];
    __shared__ __align__(16) bf16_t fbf[32][512];
    __shared__ float proj[32][128];
    int b = blockIdx.x;
    const float* tb = tin + (size_t)b * 28 * 128;
    for (int i = threadIdx.x; i < 32 * 128; i += 256) {
        int r = i >> 7, c = i & 127;
        abf[r][c] = f2bf(r < 28 ? tb[r * 128 + c] : 0.f);
    }
    __syncthreads();
    int wave = threadIdx.x >> 5, lane = threadIdx.x & 31;
    int half = lane >> 4, ln = lane & 15;
    // FF1: 2 x 32 tiles, K=128 -> gelu -> fbf
    for (int tile = wave; tile < 64; tile += 8) {
        int mt = tile >> 5, nt = tile & 31;
        v8f acc = {};
        for (int kk = 0; kk < 4; kk++) {
            v16bf a  = load_frag_A(&abf[0][0], 128, mt * 16, kk, lane);
            v16bf bv = load_frag_B(w1, 128, nt * 16 + ln, kk, lane);
            acc = wmma_bf16(a, bv, acc);
        }
        int n = nt * 16 + ln;
        float bias = b1[n];
        for (int r = 0; r < 8; r++)
            fbf[mt * 16 + half * 8 + r][n] = f2bf(geluf(acc[r] + bias));
    }
    __syncthreads();
    // FF2: 2 x 8 tiles, K=512 (16 steps)
    for (int tile = wave; tile < 16; tile += 8) {
        int mt = tile >> 3, nt = tile & 7;
        v8f acc = {};
        for (int kk = 0; kk < 16; kk++) {
            v16bf a  = load_frag_A(&fbf[0][0], 512, mt * 16, kk, lane);
            v16bf bv = load_frag_B(w2, 512, nt * 16 + ln, kk, lane);
            acc = wmma_bf16(a, bv, acc);
        }
        int n = nt * 16 + ln;
        float bias = b2[n];
        for (int r = 0; r < 8; r++) proj[mt * 16 + half * 8 + r][n] = acc[r] + bias;
    }
    __syncthreads();
    for (int r = wave; r < 28; r += 8) {
        float vals[4];
        float s = 0.f;
        for (int j = 0; j < 4; j++) {
            int c = lane * 4 + j;
            vals[j] = tb[r * 128 + c] + proj[r][c];
            s += vals[j];
        }
        for (int off = 16; off; off >>= 1) s += __shfl_xor(s, off, 32);
        float mean = s * (1.f / 128.f);
        float vs = 0.f;
        for (int j = 0; j < 4; j++) { float d = vals[j] - mean; vs += d * d; }
        for (int off = 16; off; off >>= 1) vs += __shfl_xor(vs, off, 32);
        float inv = rsqrtf(vs * (1.f / 128.f) + 1e-5f);
        float* outr = tout + (size_t)b * 28 * 128 + r * 128;
        for (int j = 0; j < 4; j++) {
            int c = lane * 4 + j;
            outr[c] = (vals[j] - mean) * inv * lng[c] + lnb[c];
        }
    }
}

// ---------------------------------------------------------------------------
// ERP features (13 per batch row) from raw x
// ---------------------------------------------------------------------------
__global__ void __launch_bounds__(256) k_erp(const float* __restrict__ x,
                                             float* __restrict__ erp) {
    __shared__ float lat[132], amp[132], n2a[132], apa[132], bpa[132];
    int b = blockIdx.x, c = threadIdx.x;
    if (c < 129) {
        const float* xc = x + ((size_t)b * 129 + c) * 900;
        float mx = -1e30f; int idx = 300;
        float mn = 1e30f;
        float ap = 0.f, bp = 0.f;
        for (int t = 0; t < 900; t++) {
            float vv = xc[t];
            if (t < 300) ap += vv * vv;
            else if (t < 600) {
                bp += vv * vv;
                if (vv > mx) { mx = vv; idx = t; }
            }
            if (t >= 180 && t < 315) mn = fminf(mn, vv);
        }
        lat[c] = (float)idx * 0.01f;   // (s+argmax)/100, t is absolute
        amp[c] = mx;
        n2a[c] = mn;
        apa[c] = ap * (1.f / 300.f);
        bpa[c] = bp * (1.f / 300.f);
    }
    __syncthreads();
    if (threadIdx.x == 0) {
        float sl = 0, sa = 0, sal = 0, amx = -1e30f, sn = 0, nmn = 1e30f,
              sap = 0, apmx = -1e30f, sbp = 0;
        for (int i = 0; i < 129; i++) {
            sl += lat[i]; sa += amp[i]; sal += amp[i] * lat[i];
            amx = fmaxf(amx, amp[i]);
            sn += n2a[i]; nmn = fminf(nmn, n2a[i]);
            sap += apa[i]; apmx = fmaxf(apmx, apa[i]);
            sbp += bpa[i];
        }
        const float inv = 1.f / 129.f, invm1 = 1.f / 128.f;
        float ml = sl * inv, mn2 = sn * inv, map = sap * inv, mbp = sbp * inv;
        float vl = 0, vn = 0, vap = 0, vbp = 0;
        for (int i = 0; i < 129; i++) {
            float d;
            d = lat[i] - ml;  vl  += d * d;
            d = n2a[i] - mn2; vn  += d * d;
            d = apa[i] - map; vap += d * d;
            d = bpa[i] - mbp; vbp += d * d;
        }
        float* o = erp + (size_t)b * 13;
        o[0]  = ml;            o[1]  = sqrtf(vl * invm1);
        o[2]  = sa * inv;      o[3]  = amx;   o[4] = sal * inv;
        o[5]  = mn2;           o[6]  = sqrtf(vn * invm1);  o[7] = nmn;
        o[8]  = map;           o[9]  = sqrtf(vap * invm1); o[10] = apmx;
        o[11] = mbp;           o[12] = sqrtf(vbp * invm1);
    }
}

// ---------------------------------------------------------------------------
// Pack fusion input: feat[b][k] = learned(d*28+s) | erp | zero-pad to 3616
// ---------------------------------------------------------------------------
__global__ void k_pack_feat(const float* __restrict__ tA,
                            const float* __restrict__ erp,
                            bf16_t* __restrict__ feat) {
    int b = blockIdx.x;
    for (int k = threadIdx.x; k < 3616; k += 256) {
        float vv;
        if (k < 3584) {
            int d = k / 28, s = k - d * 28;
            vv = tA[((size_t)b * 28 + s) * 128 + d];
        } else if (k < 3597) {
            vv = erp[(size_t)b * 13 + (k - 3584)];
        } else vv = 0.f;
        feat[(size_t)b * 3616 + k] = f2bf(vv);
    }
}

// ---------------------------------------------------------------------------
// Generic WMMA GEMM + bias + ELU, out bf16:  out = elu(A @ W^T + bias)
// A: M x Kp bf16, W: N x Kp bf16 (Kp multiple of 32), M,N multiples of 16
// ---------------------------------------------------------------------------
__global__ void __launch_bounds__(256) k_gemm_elu(const bf16_t* __restrict__ A,
        const bf16_t* __restrict__ W, const float* __restrict__ bias,
        bf16_t* __restrict__ out, int M, int N, int Kp) {
    int wave = threadIdx.x >> 5, lane = threadIdx.x & 31;
    int half = lane >> 4, ln = lane & 15;
    int Nt = N >> 4;
    int total = (M >> 4) * Nt;
    int tile = blockIdx.x * 8 + wave;
    if (tile >= total) return;
    int mt = tile / Nt, nt = tile - mt * Nt;
    int n = nt * 16 + ln;
    v8f acc = {};
    int Ki = Kp >> 5;
    for (int kk = 0; kk < Ki; kk++) {
        if (kk + 1 < Ki)
            __builtin_prefetch(W + (size_t)n * Kp + (kk + 1) * 32, 0, 1);
        v16bf a  = load_frag_A(A, Kp, mt * 16, kk, lane);
        v16bf bv = load_frag_B(W, Kp, n, kk, lane);
        acc = wmma_bf16(a, bv, acc);
    }
    float bi = bias[n];
    for (int r = 0; r < 8; r++) {
        float vv = eluf(acc[r] + bi);
        out[(size_t)(mt * 16 + half * 8 + r) * N + n] = f2bf(vv);
    }
}

// ---------------------------------------------------------------------------
// Final: y = clip(y3 @ w4^T + b4, 0.5, 1.5) -> d_out (512 f32)
// ---------------------------------------------------------------------------
__global__ void k_final(const bf16_t* __restrict__ y3, const float* w4,
                        const float* b4, float* __restrict__ out) {
    int b = blockIdx.x * 256 + threadIdx.x;
    if (b < 512) {
        float s = b4[0];
        for (int c = 0; c < 128; c++) s += bf2f(y3[(size_t)b * 128 + c]) * w4[c];
        out[b] = fminf(fmaxf(s, 0.5f), 1.5f);
    }
}

// ---------------------------------------------------------------------------
// Host launcher
// ---------------------------------------------------------------------------
extern "C" void kernel_launch(void* const* d_in, const int* in_sizes, int n_in,
                              void* d_out, int out_size, void* d_ws, size_t ws_size,
                              hipStream_t stream) {
    (void)in_sizes; (void)n_in; (void)out_size; (void)ws_size;
    int i = 0;
    const float* x    = (const float*)d_in[i++];   // (512,129,900)
    const float* sp_w = (const float*)d_in[i++];   // (32,129)
    const float* bn1g = (const float*)d_in[i++];
    const float* bn1b = (const float*)d_in[i++];
    const float* bn1m = (const float*)d_in[i++];
    const float* bn1v = (const float*)d_in[i++];
    const float* tm_w = (const float*)d_in[i++];   // (64,32,25)
    const float* bn2g = (const float*)d_in[i++];
    const float* bn2b = (const float*)d_in[i++];
    const float* bn2m = (const float*)d_in[i++];
    const float* bn2v = (const float*)d_in[i++];
    const float* s1dw = (const float*)d_in[i++];   // (64,1,5)
    const float* s1pw = (const float*)d_in[i++];   // (128,64)
    const float* s1g  = (const float*)d_in[i++];
    const float* s1b  = (const float*)d_in[i++];
    const float* s1m  = (const float*)d_in[i++];
    const float* s1v  = (const float*)d_in[i++];
    const float* s2dw = (const float*)d_in[i++];   // (128,1,5)
    const float* s2pw = (const float*)d_in[i++];   // (128,128)
    const float* s2g  = (const float*)d_in[i++];
    const float* s2b  = (const float*)d_in[i++];
    const float* s2m  = (const float*)d_in[i++];
    const float* s2v  = (const float*)d_in[i++];
    const float* Lp[4][12];   // in_w,in_b,out_w,out_b,ln1_g,ln1_b,ff1_w,ff1_b,ff2_w,ff2_b,ln2_g,ln2_b
    for (int l = 0; l < 4; l++)
        for (int j = 0; j < 12; j++) Lp[l][j] = (const float*)d_in[i++];
    const float* fw1 = (const float*)d_in[i++];    // (512,3597)
    const float* fb1 = (const float*)d_in[i++];
    const float* fw2 = (const float*)d_in[i++];    // (256,512)
    const float* fb2 = (const float*)d_in[i++];
    const float* fw3 = (const float*)d_in[i++];    // (128,256)
    const float* fb3 = (const float*)d_in[i++];
    const float* fw4 = (const float*)d_in[i++];    // (1,128)
    const float* fb4 = (const float*)d_in[i++];

    float*  wsf = (float*)d_ws;
    bf16_t* wsb = (bf16_t*)((char*)d_ws + F_END * sizeof(float));

    float* h1  = wsf + F_H1;
    float* h2  = wsf + F_H2;
    float* h3  = h1;            // reuse (h1 dead after temporal)
    float* h4  = h2;            // reuse (h2 dead after sc1)
    float* tA  = wsf + F_TA;
    float* tB  = wsf + F_TB;
    float* erp = wsf + F_ERP;

    bf16_t* feat = wsb + B_FEAT;
    bf16_t* y1   = wsb + B_Y1;
    bf16_t* y2   = wsb + B_Y2;
    bf16_t* y3   = wsb + B_Y3;
    bf16_t* tmw  = wsb + B_TMW;
    bf16_t* w1bf = wsb + B_W1;
    bf16_t* w2bf = wsb + B_W2;
    bf16_t* w3bf = wsb + B_W3;

    // per-layer bf16 weights: [in_w | out_w | ff1_w | ff2_w]
    bf16_t* lin[4]; bf16_t* lout[4]; bf16_t* lff1[4]; bf16_t* lff2[4];
    for (int l = 0; l < 4; l++) {
        bf16_t* base = wsb + B_LW + (size_t)l * LW_STRIDE;
        lin[l]  = base;
        lout[l] = base + 49152;
        lff1[l] = base + 49152 + 16384;
        lff2[l] = base + 49152 + 16384 + 65536;
    }

    // --- weight conversions (f32 -> bf16, L2-resident afterwards) ---
    k_cvt<<<64, 256, 0, stream>>>(tm_w, tmw, 64 * 800);
    for (int l = 0; l < 4; l++) {
        k_cvt<<<48, 256, 0, stream>>>(Lp[l][0], lin[l],  49152);
        k_cvt<<<16, 256, 0, stream>>>(Lp[l][2], lout[l], 16384);
        k_cvt<<<64, 256, 0, stream>>>(Lp[l][6], lff1[l], 65536);
        k_cvt<<<64, 256, 0, stream>>>(Lp[l][8], lff2[l], 65536);
    }
    k_cvt_pad<<<512, 256, 0, stream>>>(fw1, w1bf, 512, 3597, 3616);
    k_cvt<<<128, 256, 0, stream>>>(fw2, w2bf, 131072);
    k_cvt<<<32, 256, 0, stream>>>(fw3, w3bf, 32768);

    // --- CNN front-end ---
    k_spatial <<<dim3(15, 512), 256, 0, stream>>>(x, sp_w, bn1g, bn1b, bn1m, bn1v, h1);
    k_temporal<<<dim3(29, 512), 256, 0, stream>>>(h1, tmw, bn2g, bn2b, bn2m, bn2v, h2);
    k_sc1     <<<dim3(7, 512), 256, 0, stream>>>(h2, s1dw, s1pw, s1g, s1b, s1m, s1v, h3);
    k_sc2     <<<dim3(7, 512), 256, 0, stream>>>(h3, s2dw, s2pw, s2g, s2b, s2m, s2v, h4);
    k_pack_t  <<<512, 256, 0, stream>>>(h4, tA);

    // --- ERP features (independent of CNN path) ---
    k_erp<<<512, 256, 0, stream>>>(x, erp);

    // --- transformer (4 layers, ping-pong tA/tB) ---
    for (int l = 0; l < 4; l++) {
        k_attn<<<512, 256, 0, stream>>>(tA, lin[l], Lp[l][1], lout[l], Lp[l][3],
                                        Lp[l][4], Lp[l][5], tB);
        k_ffn <<<512, 256, 0, stream>>>(tB, lff1[l], Lp[l][7], lff2[l], Lp[l][9],
                                        Lp[l][10], Lp[l][11], tA);
    }

    // --- fusion MLP ---
    k_pack_feat<<<512, 256, 0, stream>>>(tA, erp, feat);
    k_gemm_elu<<<128, 256, 0, stream>>>(feat, w1bf, fb1, y1, 512, 512, 3616);
    k_gemm_elu<<<64, 256, 0, stream>>>(y1, w2bf, fb2, y2, 512, 256, 512);
    k_gemm_elu<<<32, 256, 0, stream>>>(y2, w3bf, fb3, y3, 512, 128, 256);
    k_final<<<2, 256, 0, stream>>>(y3, fw4, fb4, (float*)d_out);
}
